// TransducerJoint_72980084293741
// MI455X (gfx1250) — matrix operand order repaired and checked by reading, not compiled, not run on hardware
//
#include <hip/hip_runtime.h>

// ---------------------------------------------------------------------------
// Transducer joint: out[b,t,u,v] = relu(f[b,t,:] + g[b,u,:]) @ W_out + b_out
//   f = enc^T @ W_enc + b_enc   g = pred^T @ W_pred + b_pred
// bf16 WMMA (v_wmma_f32_16x16x32_bf16) everywhere; W pre-packed in B-fragment
// order; A tiles staged in LDS; both GEMMs double-buffer fragments in
// registers; the 537MB output is stored non-temporally (write-once).
// ---------------------------------------------------------------------------

typedef __bf16 bf16_t;
typedef __attribute__((ext_vector_type(16))) __bf16 v16bf;
typedef __attribute__((ext_vector_type(8)))  __bf16 v8bf;
typedef __attribute__((ext_vector_type(8)))  float  v8f;

constexpr int kB  = 8;
constexpr int kT  = 256;
constexpr int kU  = 64;
constexpr int kH1 = 512;
constexpr int kH2 = 640;
constexpr int kH  = 640;
constexpr int kV1 = 1025;
constexpr int kVPAD = 1280;          // V1 padded to multiple of 256
constexpr int ASTRIDE = 648;         // 640 + 8 bf16 pad (spreads LDS banks)
constexpr int kJointSmemBytes = 64 * ASTRIDE * 2 /*A tile*/ + 256 * 4 /*bias*/;

// ---------------------------------------------------------------------------
// (B, H, L) f32  ->  (B*L, H) bf16   (transpose + downconvert)
// ---------------------------------------------------------------------------
__global__ void transpose_to_bf16(const float* __restrict__ in,
                                  bf16_t* __restrict__ out,
                                  int Bn, int Hn, int Ln) {
  int total = Bn * Hn * Ln;
  for (int idx = blockIdx.x * blockDim.x + threadIdx.x; idx < total;
       idx += gridDim.x * blockDim.x) {
    int h = idx % Hn;
    int l = (idx / Hn) % Ln;
    int b = idx / (Hn * Ln);
    out[idx] = (bf16_t)in[((size_t)b * Hn + h) * Ln + l];  // coalesced writes
  }
}

// ---------------------------------------------------------------------------
// Pack W (K x N, f32) into WMMA B-fragment blocks, bf16.  Block (kt, ntb) is a
// 32x16 tile: lane l, element j ->  n = ntb*16 + (l&15), k = kt*32+8*(l>>4)+(j&7)+16*(j>>3)
// => each lane's fragment is 16 contiguous bf16 (32B), fully coalesced.
// ---------------------------------------------------------------------------
__global__ void pack_w_bf16(const float* __restrict__ W,
                            bf16_t* __restrict__ out,
                            int K, int N, int Npad) {
  int nblk = Npad / 16;
  int total = (K / 32) * nblk * 512;
  for (int idx = blockIdx.x * blockDim.x + threadIdx.x; idx < total;
       idx += gridDim.x * blockDim.x) {
    int j    = idx & 15;
    int lane = (idx >> 4) & 31;
    int blk  = idx >> 9;
    int kt   = blk / nblk;
    int ntb  = blk % nblk;
    int n = ntb * 16 + (lane & 15);
    int k = kt * 32 + 8 * (lane >> 4) + (j & 7) + 16 * (j >> 3);
    float v = (n < N) ? W[(size_t)k * N + n] : 0.0f;
    out[idx] = (bf16_t)v;
  }
}

__device__ __forceinline__ v16bf make_a_frag(const bf16_t* p) {
  // A-fragment per ISA layout: lane's 16 values are two 8-element K chunks
  // (base+0..7 and base+16..23), each a 16B load.
  v8bf a0 = *(const v8bf*)p;
  v8bf a1 = *(const v8bf*)(p + 16);
  return __builtin_shufflevector(a0, a1, 0, 1, 2, 3, 4, 5, 6, 7,
                                 8, 9, 10, 11, 12, 13, 14, 15);
}

__device__ __forceinline__ v8f wmma_bf16(v16bf a, v16bf b, v8f c) {
  return __builtin_amdgcn_wmma_f32_16x16x32_bf16(false, a, false, b, (short)0,
                                                 c, false, false);
}

// ---------------------------------------------------------------------------
// Projection GEMM: C(M, Npad) = A(M, K) @ Wpacked + bias, bf16 in / bf16 out.
// A and B fragments double-buffered in registers so v_wmma only waits on the
// PREVIOUS iteration's loads (one full K-step of latency hiding).
// ---------------------------------------------------------------------------
__global__ __launch_bounds__(128) void proj_gemm_bf16(
    const bf16_t* __restrict__ A,    // M x K, row-major
    const bf16_t* __restrict__ Bp,   // packed (K/32, Npad/16, 32, 16)
    const float*  __restrict__ bias, // length Npad
    bf16_t* __restrict__ C,          // M x Npad, row-major
    int K, int Npad) {
  const int lane = threadIdx.x & 31;
  const int wave = threadIdx.x >> 5;
  const int hlf  = lane >> 4;
  const int ntilesN = Npad / 64;
  const int tm = blockIdx.x / ntilesN;
  const int tn = blockIdx.x % ntilesN;
  const int nblk = Npad / 16;
  const int arow = tm * 64 + wave * 16 + (lane & 15);

  v8f acc[4];
#pragma unroll
  for (int jj = 0; jj < 4; ++jj)
#pragma unroll
    for (int r = 0; r < 8; ++r) acc[jj][r] = 0.0f;

  const int nSteps = K / 32;
  const bf16_t* abase = A + (size_t)arow * K + hlf * 8;
  const bf16_t* bbase = Bp + (((size_t)tn * 4) << 9) + lane * 16;
  const size_t bstep = (size_t)nblk << 9;  // elements per K-step

  v16bf a_cur = make_a_frag(abase);
  v16bf b_cur[4], b_nxt[4];
#pragma unroll
  for (int jj = 0; jj < 4; ++jj)
    b_cur[jj] = *(const v16bf*)(bbase + (size_t)jj * 512);

#pragma unroll 2
  for (int ks = 0; ks < nSteps; ++ks) {
    int ksn = ks + 1 < nSteps ? ks + 1 : ks;  // branchless prefetch
    v16bf a_nxt = make_a_frag(abase + ksn * 32);
#pragma unroll
    for (int jj = 0; jj < 4; ++jj)
      b_nxt[jj] = *(const v16bf*)(bbase + (size_t)ksn * bstep + jj * 512);
#pragma unroll
    for (int jj = 0; jj < 4; ++jj)
      acc[jj] = wmma_bf16(a_cur, b_cur[jj], acc[jj]);
    a_cur = a_nxt;
#pragma unroll
    for (int jj = 0; jj < 4; ++jj) b_cur[jj] = b_nxt[jj];
  }

  // C/D layout: m = 8*(lane>>4) + r, n = lane & 15 (within 16x16 subtile)
#pragma unroll
  for (int jj = 0; jj < 4; ++jj) {
    int n = tn * 64 + jj * 16 + (lane & 15);
    float bz = bias[n];
#pragma unroll
    for (int r = 0; r < 8; ++r) {
      int m = tm * 64 + wave * 16 + hlf * 8 + r;
      C[(size_t)m * Npad + n] = (bf16_t)(acc[jj][r] + bz);
    }
  }
}

// ---------------------------------------------------------------------------
// Joint GEMM. Block = 256 threads (8 waves), wave grid 2(M) x 4(N); each wave
// does M=32 x N=64 so every B fragment feeds 2 WMMAs from registers.
// Block handles one (b,t) (all 64 u rows) and one 256-wide N tile.
// A tile (64 x 640) = relu(f[b,t,:] + g[b,u,:]) built in dynamic LDS (83KB).
// B fragments double-buffered in registers; output stored non-temporally.
// ---------------------------------------------------------------------------
__global__ __launch_bounds__(256) void joint_gemm_bf16(
    const bf16_t* __restrict__ f,   // (B*T, H)
    const bf16_t* __restrict__ g,   // (B*U, H)
    const bf16_t* __restrict__ Wp,  // packed (H/32, VPAD/16, 32, 16)
    const float* __restrict__ b_out,
    float* __restrict__ out) {      // (B*T*U, V1)
  extern __shared__ char smem[];
  bf16_t* sA = (bf16_t*)smem;                              // 64 x ASTRIDE
  float* sBias = (float*)(smem + (size_t)64 * ASTRIDE * 2);  // 256 floats

  const int bid = blockIdx.x;
  const int nt5 = bid % 5;          // 256-wide N tile (VPAD/256 = 5)
  const int bt  = bid / 5;          // fused (b, t)
  const int bb  = bt / kT;

  const int lane = threadIdx.x & 31;
  const int wave = threadIdx.x >> 5;
  const int wm = wave & 1;          // wave M index (0..1) -> 32 rows each
  const int wn = wave >> 1;         // wave N index (0..3) -> 64 cols each
  const int hlf = lane >> 4;

  // ---- stage bias slice into LDS with a CDNA5 async global->LDS copy ----
  {
    int v = nt5 * 256 + (int)threadIdx.x;
    int vc = v < kV1 ? v : kV1 - 1;
    unsigned ldsOff = (unsigned)(64 * ASTRIDE * 2) + threadIdx.x * 4u;
    unsigned long long ga = (unsigned long long)(b_out + vc);
    asm volatile("global_load_async_to_lds_b32 %0, %1, off"
                 :: "v"(ldsOff), "v"(ga) : "memory");
  }

  // ---- build A tile: relu(f + g), 64 rows x 640, 16B vector chunks ----
  {
    const bf16_t* frow = f + (size_t)bt * kH;
    const int chunks = kH / 8;  // 80
    for (int task = threadIdx.x; task < 64 * chunks; task += blockDim.x) {
      int c8 = task % chunks;
      int u  = task / chunks;
      v8bf fv = *(const v8bf*)(frow + c8 * 8);
      v8bf gv = *(const v8bf*)(g + ((size_t)(bb * kU + u)) * kH + c8 * 8);
      v8bf hv;
#pragma unroll
      for (int e = 0; e < 8; ++e) {
        float s = (float)fv[e] + (float)gv[e];
        hv[e] = (bf16_t)(s > 0.0f ? s : 0.0f);
      }
      *(v8bf*)(&sA[u * ASTRIDE + c8 * 8]) = hv;
    }
  }
  asm volatile("s_wait_asynccnt 0x0" ::: "memory");
  __syncthreads();

  // ---- K loop: 20 steps, 8 WMMAs/step, B double-buffered in registers ----
  v8f acc[2][4];
#pragma unroll
  for (int mi = 0; mi < 2; ++mi)
#pragma unroll
    for (int jj = 0; jj < 4; ++jj)
#pragma unroll
      for (int r = 0; r < 8; ++r) acc[mi][jj][r] = 0.0f;

  const int nblk = kVPAD / 16;  // 80
  const bf16_t* bbase =
      Wp + (((size_t)nt5 * 16 + wn * 4) << 9) + lane * 16;
  const size_t bstep = (size_t)nblk << 9;  // elements per K-step
  const int a0row = wm * 32 + (lane & 15);
  const int nSteps = kH / 32;  // 20

  v16bf bcur[4], bnxt[4];
#pragma unroll
  for (int jj = 0; jj < 4; ++jj)
    bcur[jj] = *(const v16bf*)(bbase + (size_t)jj * 512);

#pragma unroll 2
  for (int ks = 0; ks < nSteps; ++ks) {
    int ksn = ks + 1 < nSteps ? ks + 1 : ks;  // branchless prefetch
#pragma unroll
    for (int jj = 0; jj < 4; ++jj)
      bnxt[jj] = *(const v16bf*)(bbase + (size_t)ksn * bstep + jj * 512);
    v16bf a0 = make_a_frag(&sA[a0row * ASTRIDE + ks * 32 + hlf * 8]);
    v16bf a1 = make_a_frag(&sA[(a0row + 16) * ASTRIDE + ks * 32 + hlf * 8]);
#pragma unroll
    for (int jj = 0; jj < 4; ++jj) {
      acc[0][jj] = wmma_bf16(a0, bcur[jj], acc[0][jj]);
      acc[1][jj] = wmma_bf16(a1, bcur[jj], acc[1][jj]);
    }
#pragma unroll
    for (int jj = 0; jj < 4; ++jj) bcur[jj] = bnxt[jj];
  }

  // ---- epilogue: bias from LDS + guarded non-temporal fp32 stores ----
#pragma unroll
  for (int jj = 0; jj < 4; ++jj) {
    int vloc = wn * 64 + jj * 16 + (lane & 15);
    int v = nt5 * 256 + vloc;
    if (v < kV1) {
      float bz = sBias[vloc];
#pragma unroll
      for (int mi = 0; mi < 2; ++mi) {
        int ubase = wm * 32 + mi * 16 + hlf * 8;
#pragma unroll
        for (int r = 0; r < 8; ++r) {
          __builtin_nontemporal_store(
              acc[mi][jj][r] + bz,
              &out[((size_t)bt * kU + (ubase + r)) * kV1 + v]);
        }
      }
    }
  }
}

// ---------------------------------------------------------------------------
extern "C" void kernel_launch(void* const* d_in, const int* in_sizes, int n_in,
                              void* d_out, int out_size, void* d_ws,
                              size_t ws_size, hipStream_t stream) {
  const float* enc    = (const float*)d_in[0];  // (B, H1, T)
  const float* pred   = (const float*)d_in[1];  // (B, H2, U)
  const float* W_enc  = (const float*)d_in[2];  // (H1, H)
  const float* b_enc  = (const float*)d_in[3];  // (H,)
  const float* W_pred = (const float*)d_in[4];  // (H2, H)
  const float* b_pred = (const float*)d_in[5];  // (H,)
  const float* W_out  = (const float*)d_in[6];  // (H, V1)
  const float* b_out  = (const float*)d_in[7];  // (V1,)
  float* out = (float*)d_out;

  // carve workspace (256B aligned chunks)
  char* ws = (char*)d_ws;
  auto carve = [&](size_t bytes) {
    char* p = ws;
    ws += (bytes + 255) & ~(size_t)255;
    return p;
  };
  bf16_t* encT   = (bf16_t*)carve((size_t)kB * kT * kH1 * 2);
  bf16_t* predT  = (bf16_t*)carve((size_t)kB * kU * kH2 * 2);
  bf16_t* WencP  = (bf16_t*)carve((size_t)(kH1 / 32) * (kH / 16) * 512 * 2);
  bf16_t* WpredP = (bf16_t*)carve((size_t)(kH2 / 32) * (kH / 16) * 512 * 2);
  bf16_t* WoutP  = (bf16_t*)carve((size_t)(kH / 32) * (kVPAD / 16) * 512 * 2);
  bf16_t* fbuf   = (bf16_t*)carve((size_t)kB * kT * kH * 2);
  bf16_t* gbuf   = (bf16_t*)carve((size_t)kB * kU * kH * 2);
  (void)ws_size; (void)in_sizes; (void)n_in; (void)out_size;

  // allow 83KB dynamic LDS for the joint kernel (idempotent, capture-safe)
  (void)hipFuncSetAttribute((const void*)joint_gemm_bf16,
                            hipFuncAttributeMaxDynamicSharedMemorySize,
                            kJointSmemBytes);

  // 1) transpose + downconvert activations
  transpose_to_bf16<<<256, 256, 0, stream>>>(enc, encT, kB, kH1, kT);
  transpose_to_bf16<<<256, 256, 0, stream>>>(pred, predT, kB, kH2, kU);

  // 2) pack weights into WMMA B-fragment order (bf16)
  pack_w_bf16<<<256, 256, 0, stream>>>(W_enc, WencP, kH1, kH, kH);
  pack_w_bf16<<<256, 256, 0, stream>>>(W_pred, WpredP, kH2, kH, kH);
  pack_w_bf16<<<512, 256, 0, stream>>>(W_out, WoutP, kH, kV1, kVPAD);

  // 3) projections (bias fused), bf16 outputs
  proj_gemm_bf16<<<(kB * kT / 64) * (kH / 64), 128, 0, stream>>>(
      encT, WencP, b_enc, fbuf, kH1, kH);
  proj_gemm_bf16<<<(kB * kU / 64) * (kH / 64), 128, 0, stream>>>(
      predT, WpredP, b_pred, gbuf, kH2, kH);

  // 4) joint GEMM: relu(f+g) @ W_out + b_out  (dominant, store-bound)
  joint_gemm_bf16<<<kB * kT * 5, 256, kJointSmemBytes, stream>>>(
      fbuf, gbuf, WoutP, b_out, out);
}
